// AttentionElement_42202348650568
// MI455X (gfx1250) — matrix operand order repaired and checked by reading, not compiled, not run on hardware
//
#include <hip/hip_runtime.h>
#include <hip/hip_fp16.h>

// ---------------------------------------------------------------------------
// Fused neighborhood attention for MI455X (gfx1250, wave32, WMMA).
//
// Algebraic refactoring (exact w.r.t. the reference math):
//   qk[n,m]  = CE[n,:] @ (Wq @ Wk^T)[:,m] + (Wk @ bq)[m]
//   logits[n,k] = qk[n,0:192] . rel[k,:] + qk[n,192:256] . spatial[n,k,:]
//                 (+ query.bk is constant over k -> softmax invariant, dropped)
//   scores = softmax(logits - (1-mask)*1e9)
//   mbar[n,0:192]   = scores[n,:] @ rel
//   mbar[n,192:256] = scores[n,:] @ spatial[n]
//   out = mbar @ (Wv @ Wo) + (Wo^T bv + bo)     (since sum_k scores == 1)
// ---------------------------------------------------------------------------

typedef _Float16 v16h __attribute__((ext_vector_type(16)));
typedef _Float16 v8h  __attribute__((ext_vector_type(8)));
typedef float    v8f  __attribute__((ext_vector_type(8)));

#define NKPTS 343      // 7*7*7
#define KPAD  352      // padded to 22*16
#define MCH   256      // memory channels
#define RCH   192      // rel channels
#define SCH   64       // spatial channels
#define TILE_N 16      // n-rows per block

// ---- WMMA fragment helpers (contiguous k order, shared by A and B) -----------------

__device__ __forceinline__ v8f wmma16(v16h a, v16h b, v8f c) {
  // D = A(16x32 f16) * B(32x16 f16) + C(16x16 f32)
  return __builtin_amdgcn_wmma_f32_16x16x32_f16(
      false, a, false, b, (short)0, c, false, false);
}

// A fragment: row-major f16 [16 x strideH] in LDS, K-chunk base kc.
// lane: m = lane&15, g = lane>>4; a[h] = A[m][kc + 16g + h]
__device__ __forceinline__ v16h load_a_frag(const _Float16* As, int strideH,
                                            int kc, int lane) {
  int m = lane & 15, g = lane >> 4;
  const _Float16* p = As + m * strideH + kc + 16 * g;
  v8h lo = *(const v8h*)p;
  v8h hi = *(const v8h*)(p + 8);
  v16h a;
#pragma unroll
  for (int i = 0; i < 8; ++i) { a[i] = lo[i]; a[i + 8] = hi[i]; }
  return a;
}

// B fragment: B stored transposed [N][K] f16 in global; lane n = lane&15,
// g = lane>>4; b[h] = B[kc + 16g + h][n] -> 32 contiguous bytes per lane.
__device__ __forceinline__ v16h load_b_frag(const _Float16* Bg, int strideH,
                                            int n0, int kc, int lane) {
  int nn = lane & 15, g = lane >> 4;
  return *(const v16h*)(Bg + (size_t)(n0 + nn) * strideH + kc + 16 * g);
}

// ---------------------------------------------------------------------------
// Precompute kernel: fused weights/biases + rel tables into workspace.
// ---------------------------------------------------------------------------
__global__ __launch_bounds__(256) void precompute_kernel(
    const float* __restrict__ Wq, const float* __restrict__ bq,
    const float* __restrict__ Wk, const float* __restrict__ Wv,
    const float* __restrict__ bv, const float* __restrict__ Wo,
    const float* __restrict__ bo, const float* __restrict__ sdr,
    _Float16* __restrict__ WqkT, _Float16* __restrict__ WvoT,
    _Float16* __restrict__ relB, _Float16* __restrict__ relB2,
    float* __restrict__ bqk, float* __restrict__ bvo) {
  const int total = 65536 + 65536 + 256 + 256 + KPAD * RCH + RCH * KPAD;
  for (int idx = blockIdx.x * blockDim.x + threadIdx.x; idx < total;
       idx += gridDim.x * blockDim.x) {
    int t = idx;
    if (t < 65536) {  // WqkT[m][j] = sum_c Wq[j,c] * Wk[m,c]
      int m = t >> 8, j = t & 255;
      float s = 0.f;
      for (int c = 0; c < 256; ++c) s += Wq[j * 256 + c] * Wk[m * 256 + c];
      WqkT[t] = (_Float16)s;
      continue;
    }
    t -= 65536;
    if (t < 65536) {  // WvoT[f][m] = sum_o Wv[m,o] * Wo[o,f]
      int f = t >> 8, m = t & 255;
      float s = 0.f;
      for (int o = 0; o < 256; ++o) s += Wv[m * 256 + o] * Wo[o * 256 + f];
      WvoT[t] = (_Float16)s;
      continue;
    }
    t -= 65536;
    if (t < 256) {  // bqk[m] = sum_c bq[c] * Wk[m,c]
      float s = 0.f;
      for (int c = 0; c < 256; ++c) s += bq[c] * Wk[t * 256 + c];
      bqk[t] = s;
      continue;
    }
    t -= 256;
    if (t < 256) {  // bvo[f] = sum_o bv[o] * Wo[o,f] + bo[f]
      float s = 0.f;
      for (int o = 0; o < 256; ++o) s += bv[o] * Wo[o * 256 + t];
      bvo[t] = s + bo[t];
      continue;
    }
    t -= 256;
    if (t < KPAD * RCH) {  // relB[k][c] (stage-2 B layout: [n=k][k=c])
      int k = t / RCH, c = t % RCH;
      _Float16 v = (_Float16)0.f;
      if (k < NKPTS) {
        int x = k / 49, y = (k / 7) % 7, z = k % 7;
        int axis = c >> 6, cc = c & 63;
        int w = (axis == 0) ? x : ((axis == 1) ? y : z);
        v = (_Float16)sdr[w * 64 + cc];
      }
      relB[t] = v;
      continue;
    }
    t -= KPAD * RCH;
    {  // relB2[c][k] (stage-4 B layout: [n=c][k=k]), zero-padded k>=343
      int c = t / KPAD, k = t % KPAD;
      _Float16 v = (_Float16)0.f;
      if (k < NKPTS) {
        int x = k / 49, y = (k / 7) % 7, z = k % 7;
        int axis = c >> 6, cc = c & 63;
        int w = (axis == 0) ? x : ((axis == 1) ? y : z);
        v = (_Float16)sdr[w * 64 + cc];
      }
      relB2[t] = v;
    }
  }
}

// ---------------------------------------------------------------------------
// Main fused attention kernel: one block = 16 rows of n, 256 threads (8 waves)
// ---------------------------------------------------------------------------
__global__ __launch_bounds__(256) void attn_kernel(
    const float* __restrict__ CE, const float* __restrict__ spatial,
    const float* __restrict__ mask, const _Float16* __restrict__ WqkT,
    const _Float16* __restrict__ WvoT, const _Float16* __restrict__ relB,
    const _Float16* __restrict__ relB2, const float* __restrict__ bqk,
    const float* __restrict__ bvo, float* __restrict__ out) {
  __shared__ __align__(16) _Float16 sCE[TILE_N * MCH];    // stage1 A
  __shared__ __align__(16) _Float16 sQK[TILE_N * MCH];    // stage2 A (f16)
  __shared__ __align__(16) float    sQKsp[TILE_N * SCH];  // qk spatial chans
  __shared__ __align__(16) float    sL[TILE_N * KPAD];    // logits -> scores
  __shared__ __align__(16) _Float16 sSC[TILE_N * KPAD];   // scores f16 (A)
  __shared__ __align__(16) _Float16 sMB[TILE_N * MCH];    // mbar f16 (A)

  const int tid = threadIdx.x;
  const int wave = tid >> 5;
  const int lane = tid & 31;
  const int nbase = blockIdx.x * TILE_N;
  const int nn = lane & 15, g = lane >> 4;

  // ---- stage 0: stage CE tile into LDS as f16 -----------------------------
#pragma unroll
  for (int i = 0; i < 16; ++i) {
    int idx = tid + 256 * i;
    sCE[idx] = (_Float16)CE[(size_t)nbase * MCH + idx];
  }
  __syncthreads();

  // ---- stage 1: qk = CE @ WqkT^T + bqk  (WMMA, 2 N-tiles per wave) --------
  {
    int n0 = wave * 32;
    v8f c0 = {}, c1 = {};
#pragma unroll
    for (int kc = 0; kc < MCH; kc += 32) {
      v16h a = load_a_frag(sCE, MCH, kc, lane);
      v16h b0 = load_b_frag(WqkT, MCH, n0, kc, lane);
      v16h b1 = load_b_frag(WqkT, MCH, n0 + 16, kc, lane);
      c0 = wmma16(a, b0, c0);
      c1 = wmma16(a, b1, c1);
    }
#pragma unroll
    for (int r = 0; r < 8; ++r) {
      int m = r + 8 * g;
      int col0 = n0 + nn, col1 = n0 + 16 + nn;
      float v0 = c0[r] + bqk[col0];
      float v1 = c1[r] + bqk[col1];
      sQK[m * MCH + col0] = (_Float16)v0;
      sQK[m * MCH + col1] = (_Float16)v1;
      if (col0 >= RCH) sQKsp[m * SCH + col0 - RCH] = v0;
      if (col1 >= RCH) sQKsp[m * SCH + col1 - RCH] = v1;
    }
  }
  __syncthreads();

  // ---- stage 2: L1 = qk[:,0:192] @ rel^T  (WMMA, 22 N-tiles) --------------
  for (int nt = wave; nt < KPAD / 16; nt += 8) {
    v8f c = {};
#pragma unroll
    for (int kc = 0; kc < RCH; kc += 32) {
      v16h a = load_a_frag(sQK, MCH, kc, lane);
      v16h b = load_b_frag(relB, RCH, nt * 16, kc, lane);
      c = wmma16(a, b, c);
    }
#pragma unroll
    for (int r = 0; r < 8; ++r) {
      int m = r + 8 * g;
      sL[m * KPAD + nt * 16 + nn] = c[r];
    }
  }
  __syncthreads();

  // ---- stage 3: spatial logits + mask + softmax (2 n's per wave) ----------
  for (int rep = 0; rep < 2; ++rep) {
    const int nloc = wave * 2 + rep;
    const int ng = nbase + nloc;
    const float* spn = spatial + (size_t)ng * NKPTS * SCH;
    float acc[11];
#pragma unroll
    for (int i = 0; i < 11; ++i) acc[i] = 0.f;
#pragma unroll
    for (int ch = 0; ch < 4; ++ch) {
      float q[16];
#pragma unroll
      for (int j = 0; j < 16; ++j) q[j] = sQKsp[nloc * SCH + ch * 16 + j];
#pragma unroll
      for (int i = 0; i < 11; ++i) {
        int kk = lane + 32 * i;
        if (kk < NKPTS) {
          const float4* sp = (const float4*)(spn + (size_t)kk * SCH + ch * 16);
          float4 s0 = sp[0], s1 = sp[1], s2 = sp[2], s3 = sp[3];
          float a = acc[i];
          a += s0.x * q[0] + s0.y * q[1] + s0.z * q[2] + s0.w * q[3];
          a += s1.x * q[4] + s1.y * q[5] + s1.z * q[6] + s1.w * q[7];
          a += s2.x * q[8] + s2.y * q[9] + s2.z * q[10] + s2.w * q[11];
          a += s3.x * q[12] + s3.y * q[13] + s3.z * q[14] + s3.w * q[15];
          acc[i] = a;
        }
      }
    }
    // combine with rel logits + mask penalty
    float lmax = -3.0e38f;
#pragma unroll
    for (int i = 0; i < 11; ++i) {
      int kk = lane + 32 * i;
      if (kk < NKPTS) {
        float v = acc[i] + sL[nloc * KPAD + kk] -
                  (1.0f - mask[(size_t)ng * NKPTS + kk]) * 1.0e9f;
        acc[i] = v;
        lmax = fmaxf(lmax, v);
      }
    }
#pragma unroll
    for (int off = 16; off >= 1; off >>= 1)
      lmax = fmaxf(lmax, __shfl_xor(lmax, off));
    float lsum = 0.f;
#pragma unroll
    for (int i = 0; i < 11; ++i) {
      int kk = lane + 32 * i;
      if (kk < NKPTS) {
        acc[i] = __expf(acc[i] - lmax);
        lsum += acc[i];
      }
    }
#pragma unroll
    for (int off = 16; off >= 1; off >>= 1) lsum += __shfl_xor(lsum, off);
    float inv = 1.0f / lsum;
#pragma unroll
    for (int i = 0; i < 11; ++i) {
      int kk = lane + 32 * i;
      if (kk < NKPTS) {
        float s = acc[i] * inv;
        sL[nloc * KPAD + kk] = s;            // f32 scores (stage 5)
        sSC[nloc * KPAD + kk] = (_Float16)s; // f16 scores (stage 4 A)
      }
    }
    if (lane < KPAD - NKPTS) sSC[nloc * KPAD + NKPTS + lane] = (_Float16)0.f;
  }
  __syncthreads();

  // ---- stage 4: mbar[:,0:192] = scores @ rel  (WMMA, 12 N-tiles) ----------
  for (int nt = wave; nt < RCH / 16; nt += 8) {
    v8f c = {};
#pragma unroll
    for (int kc = 0; kc < KPAD; kc += 32) {
      v16h a = load_a_frag(sSC, KPAD, kc, lane);
      v16h b = load_b_frag(relB2, KPAD, nt * 16, kc, lane);
      c = wmma16(a, b, c);
    }
#pragma unroll
    for (int r = 0; r < 8; ++r) {
      int m = r + 8 * g;
      sMB[m * MCH + nt * 16 + nn] = (_Float16)c[r];
    }
  }

  // ---- stage 5: mbar[:,192:256] = scores @ spatial (2 n's per wave) -------
  for (int rep = 0; rep < 2; ++rep) {
    const int nloc = wave * 2 + rep;
    const int ng = nbase + nloc;
    const float* spn = spatial + (size_t)ng * NKPTS * SCH + lane * 2;
    float a0 = 0.f, a1 = 0.f;
    for (int k = 0; k < NKPTS; ++k) {
      float s = sL[nloc * KPAD + k];  // LDS broadcast read
      float2 v = *(const float2*)(spn + (size_t)k * SCH);
      a0 += s * v.x;
      a1 += s * v.y;
    }
    sMB[nloc * MCH + RCH + lane * 2] = (_Float16)a0;
    sMB[nloc * MCH + RCH + lane * 2 + 1] = (_Float16)a1;
  }
  __syncthreads();

  // ---- stage 6: out = mbar @ WvoT^T + bvo  (WMMA, 2 N-tiles per wave) -----
  {
    int n0 = wave * 32;
    v8f c0 = {}, c1 = {};
#pragma unroll
    for (int kc = 0; kc < MCH; kc += 32) {
      v16h a = load_a_frag(sMB, MCH, kc, lane);
      v16h b0 = load_b_frag(WvoT, MCH, n0, kc, lane);
      v16h b1 = load_b_frag(WvoT, MCH, n0 + 16, kc, lane);
      c0 = wmma16(a, b0, c0);
      c1 = wmma16(a, b1, c1);
    }
#pragma unroll
    for (int r = 0; r < 8; ++r) {
      int m = r + 8 * g;
      int col0 = n0 + nn, col1 = n0 + 16 + nn;
      out[(size_t)(nbase + m) * MCH + col0] = c0[r] + bvo[col0];
      out[(size_t)(nbase + m) * MCH + col1] = c1[r] + bvo[col1];
    }
  }
}

// ---------------------------------------------------------------------------
// Launch: precompute fused weights into d_ws, then fused attention.
// ---------------------------------------------------------------------------
extern "C" void kernel_launch(void* const* d_in, const int* in_sizes, int n_in,
                              void* d_out, int out_size, void* d_ws,
                              size_t ws_size, hipStream_t stream) {
  const float* CE      = (const float*)d_in[0];   // [N,256]
  const float* spatial = (const float*)d_in[1];   // [N,7,7,7,64]
  const float* mask    = (const float*)d_in[2];   // [N,343]
  const float* sdr     = (const float*)d_in[3];   // [7,64]
  const float* Wq      = (const float*)d_in[4];   // [256,256]
  const float* bq      = (const float*)d_in[5];   // [256]
  const float* Wk      = (const float*)d_in[6];   // [256,256]
  // d_in[7] = bk: adds a k-uniform constant to logits -> softmax invariant.
  const float* Wv      = (const float*)d_in[8];   // [256,256]
  const float* bv      = (const float*)d_in[9];   // [256]
  const float* Wo      = (const float*)d_in[10];  // [256,256]
  const float* bo      = (const float*)d_in[11];  // [256]
  float* out = (float*)d_out;

  const int n = in_sizes[0] / 256;  // 2048

  // workspace layout (all offsets 256B-aligned; total ~522 KB)
  char* ws = (char*)d_ws;
  _Float16* WqkT  = (_Float16*)(ws + 0);        // 256*256 f16 = 128 KB
  _Float16* WvoT  = (_Float16*)(ws + 131072);   // 128 KB
  _Float16* relB  = (_Float16*)(ws + 262144);   // 352*192 f16 = 132 KB
  _Float16* relB2 = (_Float16*)(ws + 397312);   // 192*352 f16 = 132 KB
  float*    bqk   = (float*)(ws + 532480);      // 1 KB
  float*    bvo   = (float*)(ws + 533504);      // 1 KB

  precompute_kernel<<<256, 256, 0, stream>>>(Wq, bq, Wk, Wv, bv, Wo, bo, sdr,
                                             WqkT, WvoT, relB, relB2, bqk, bvo);
  attn_kernel<<<n / TILE_N, 256, 0, stream>>>(CE, spatial, mask, WqkT, WvoT,
                                              relB, relB2, bqk, bvo, out);
}
